// Encoder_5480378270324
// MI455X (gfx1250) — matrix operand order repaired and checked by reading, not compiled, not run on hardware
//
#include <hip/hip_runtime.h>
#include <stdint.h>

// ---------------------------------------------------------------------------
// CDNA5 (gfx1250) WMMA types
// ---------------------------------------------------------------------------
typedef __attribute__((ext_vector_type(16))) __bf16        v16bf;
typedef __attribute__((ext_vector_type(8)))  float         v8f;
typedef __attribute__((ext_vector_type(8)))  unsigned int  u32x8;
typedef int v4i __attribute__((vector_size(16)));
typedef __attribute__((address_space(1))) v4i gv4i;   // global v4i
typedef __attribute__((address_space(3))) v4i lv4i;   // LDS v4i

// Async global->LDS copy support (gfx1250 GLOBAL_LOAD_ASYNC_TO_LDS_B128).
#if defined(__gfx1250__) && __has_builtin(__builtin_amdgcn_global_load_async_to_lds_b128)
#define HAS_ASYNC_LDS 1
#else
#define HAS_ASYNC_LDS 0
#endif

// Copy 16 bytes global -> LDS. Async (ASYNCcnt-tracked, no VGPR return) when
// available, otherwise load-to-reg + ds_store fallback.
__device__ __forceinline__ void copy16_g2l(const void* g, void* l) {
#if HAS_ASYNC_LDS
    __builtin_amdgcn_global_load_async_to_lds_b128(
        (gv4i*)g, (lv4i*)l, 0, 0);
#else
    *(float4*)l = *(const float4*)g;
#endif
}

__device__ __forceinline__ void wait_async_le3() {
#if HAS_ASYNC_LDS
#if __has_builtin(__builtin_amdgcn_s_wait_asynccnt)
    __builtin_amdgcn_s_wait_asynccnt(3);
#else
    asm volatile("s_wait_asynccnt 0x3" ::: "memory");
#endif
#endif
}
__device__ __forceinline__ void wait_async_le0() {
#if HAS_ASYNC_LDS
#if __has_builtin(__builtin_amdgcn_s_wait_asynccnt)
    __builtin_amdgcn_s_wait_asynccnt(0);
#else
    asm volatile("s_wait_asynccnt 0x0" ::: "memory");
#endif
#endif
}

// Round-to-nearest-even f32 -> bf16, packed two-at-a-time into a DWORD.
__device__ __forceinline__ unsigned pack_bf16x2(float lo, float hi) {
    unsigned a = __builtin_bit_cast(unsigned, lo);
    unsigned b = __builtin_bit_cast(unsigned, hi);
    a += 0x7FFFu + ((a >> 16) & 1u);
    b += 0x7FFFu + ((b >> 16) & 1u);
    return (a >> 16) | (b & 0xFFFF0000u);
}

// ---------------------------------------------------------------------------
// Pack f32 [M,512] row-major into A-fragment bf16 tiles (16x32 per tile).
// Fragment layout (ISA 7.12.2, 16-bit A 16x32): lane&15 = row, lane>>4 picks
// kbase +0/+8; elements 0..7 -> k=kb..kb+7, elements 8..15 -> k=kb+16..kb+23.
// Optional bias-add and ReLU fused (GCN epilogue).
// Tile storage: tile*512 + lane*16 uint16 -> each lane's 16 bf16 contiguous,
// each tile a contiguous 1KB block (async-copy friendly).
// ---------------------------------------------------------------------------
__global__ __launch_bounds__(256) void pack_a_kernel(
    const float* __restrict__ src, uint16_t* __restrict__ dst,
    const float* __restrict__ bias, int do_relu, int mtiles) {
    const int KT = 16;                         // K/32 with K = 512
    int t    = blockIdx.x * 256 + threadIdx.x;
    int lane = t & 31;
    int tile = t >> 5;
    if (tile >= mtiles * KT) return;
    int mtile = tile >> 4;                     // / KT
    int kt    = tile & 15;
    int row   = (mtile << 4) + (lane & 15);
    int kb    = (kt << 5) + ((lane >> 4) << 3);
    const float* p = src + (size_t)row * 512 + kb;

    float v[16];
#pragma unroll
    for (int j = 0; j < 8; ++j) v[j]     = p[j];
#pragma unroll
    for (int j = 0; j < 8; ++j) v[8 + j] = p[16 + j];
    if (bias) {
#pragma unroll
        for (int j = 0; j < 8; ++j) v[j]     += bias[kb + j];
#pragma unroll
        for (int j = 0; j < 8; ++j) v[8 + j] += bias[kb + 16 + j];
    }
    if (do_relu) {
#pragma unroll
        for (int j = 0; j < 16; ++j) v[j] = fmaxf(v[j], 0.f);
    }
    u32x8 o;
#pragma unroll
    for (int j = 0; j < 8; ++j) o[j] = pack_bf16x2(v[2 * j], v[2 * j + 1]);
    *(u32x8*)(dst + (size_t)tile * 512 + lane * 16) = o;
}

// ---------------------------------------------------------------------------
// Pack f32 weight W[512,512] (k-major rows) into B-fragment bf16 tiles
// (32x16 per tile): lane&15 = output column n, kb = kt*32 + (lane>>4)*8;
// elements 0..7 -> k=kb..kb+7, elements 8..15 -> k=kb+16..kb+23.
// ---------------------------------------------------------------------------
__global__ __launch_bounds__(256) void pack_b_kernel(
    const float* __restrict__ W, uint16_t* __restrict__ dst) {
    int t    = blockIdx.x * 256 + threadIdx.x;  // 512 tiles * 32 lanes
    int lane = t & 31;
    int tile = t >> 5;
    if (tile >= 512) return;
    int ntile = tile >> 4;
    int kt    = tile & 15;
    int n     = (ntile << 4) + (lane & 15);
    int kb    = (kt << 5) + ((lane >> 4) << 3);
    float v[16];
#pragma unroll
    for (int j = 0; j < 16; ++j) v[j] = W[(size_t)(kb + j + ((j >> 3) << 3)) * 512 + n];
    u32x8 o;
#pragma unroll
    for (int j = 0; j < 8; ++j) o[j] = pack_bf16x2(v[2 * j], v[2 * j + 1]);
    *(u32x8*)(dst + (size_t)tile * 512 + lane * 16) = o;
}

// ---------------------------------------------------------------------------
// GEMM: C[M,Ncols] = act(Apack @ Bpack + bias [+ addend]).  K = 512 fixed.
// 256 threads = 8 waves in 2(M) x 4(N); each wave owns 4 vertical 16x16 tiles
// -> block tile 128x64.  Per K-step the block's 12KB operand slice
// (8 A-tiles + 4 B-tiles, 1KB each) is double-buffered through LDS using
// async global->LDS b128 copies (3 per thread), then fragments are served to
// v_wmma_f32_16x16x32_bf16 by ds_load_b128 with zero cross-wave duplication.
// All branches are wave-uniform: EXEC stays all-ones as WMMA requires.
// ---------------------------------------------------------------------------
__global__ __launch_bounds__(256) void gemm_bf16_wmma(
    const uint16_t* __restrict__ Ap, const uint16_t* __restrict__ Bp,
    float* __restrict__ C, const float* __restrict__ bias,
    const float* __restrict__ addend, int do_relu, int Ncols) {
    const int KT = 16;
    __shared__ __align__(16) uint8_t smem[2][12288];  // [stage][8KB A | 4KB B]

    int tid  = threadIdx.x;
    int lane = tid & 31;
    int wave = tid >> 5;
    int wm   = wave >> 2;
    int wn   = wave & 3;
    int mblk = blockIdx.x * 8;    // first mtile of block
    int nblk = blockIdx.y * 4;    // first ntile of block

    const uint8_t* Abytes = (const uint8_t*)Ap;
    const uint8_t* Bbytes = (const uint8_t*)Bp;

    // Copy plan: chunk c (16B), c = tid -> A tiles 0..3, tid+256 -> A tiles
    // 4..7, tid+512 -> B tiles 0..3.  LDS dest offset == c*16.
    int tA0 = tid >> 6;                    // 0..3
    int sub = (tid & 63) << 4;             // byte offset within 1KB tile
    // kt=0 global bases for this thread's three chunks:
    const uint8_t* gA0 = Abytes + ((size_t)(mblk + tA0)     * KT) * 1024 + sub;
    const uint8_t* gA1 = Abytes + ((size_t)(mblk + tA0 + 4) * KT) * 1024 + sub;
    const uint8_t* gB  = Bbytes + ((size_t)(nblk + tA0)     * KT) * 1024 + sub;
    int l0 = tid * 16;
    int l1 = 4096 + tid * 16;
    int l2 = 8192 + tid * 16;

    // LDS fragment offsets for this wave.
    int aoff[4];
#pragma unroll
    for (int i = 0; i < 4; ++i) aoff[i] = (wm * 4 + i) * 1024 + lane * 32;
    int boff = 8192 + wn * 1024 + lane * 32;

    v8f acc[4] = {};

    // Preload stage 0.
    copy16_g2l(gA0, &smem[0][l0]);
    copy16_g2l(gA1, &smem[0][l1]);
    copy16_g2l(gB,  &smem[0][l2]);

    for (int kt = 0; kt < KT; ++kt) {
        int buf = kt & 1;
        if (kt + 1 < KT) {
            // Other buffer's previous readers were fenced by the barrier at
            // the end of iteration kt-1, so it is safe to overwrite.
            size_t adv = (size_t)(kt + 1) * 1024;
            copy16_g2l(gA0 + adv, &smem[buf ^ 1][l0]);
            copy16_g2l(gA1 + adv, &smem[buf ^ 1][l1]);
            copy16_g2l(gB  + adv, &smem[buf ^ 1][l2]);
            wait_async_le3();   // in-order completion => stage kt has landed
        } else {
            wait_async_le0();
        }
        __syncthreads();        // stage kt visible to every wave

        u32x8 braw = *(const u32x8*)&smem[buf][boff];
        v16bf bv = __builtin_bit_cast(v16bf, braw);
#pragma unroll
        for (int i = 0; i < 4; ++i) {
            u32x8 araw = *(const u32x8*)&smem[buf][aoff[i]];
            v16bf av = __builtin_bit_cast(v16bf, araw);
            acc[i] = __builtin_amdgcn_wmma_f32_16x16x32_bf16(
                false, av, false, bv, (short)0, acc[i], false, false);
        }
        __syncthreads();        // all waves done with buf before reuse
    }

    // C fragment: VGPR j, lanes 0-15 -> M=j, lanes 16-31 -> M=8+j; N = lane&15.
    int lcol  = lane & 15;
    int lhalf = lane >> 4;
    int col   = ((nblk + wn) << 4) + lcol;
    float bcol = bias ? bias[col] : 0.f;
#pragma unroll
    for (int i = 0; i < 4; ++i) {
        int rbase = ((mblk + wm * 4 + i) << 4) + (lhalf << 3);
#pragma unroll
        for (int j = 0; j < 8; ++j) {
            size_t idx = (size_t)(rbase + j) * Ncols + col;
            float v = acc[i][j] + bcol;
            if (addend) v += addend[idx];
            if (do_relu) v = fmaxf(v, 0.f);
            C[idx] = v;   // 16 lanes -> one contiguous 64B row segment
        }
    }
}

// ---------------------------------------------------------------------------
// Zero-fill (agg buffers must start at 0 before atomic aggregation).
// ---------------------------------------------------------------------------
__global__ __launch_bounds__(256) void zero_kernel(float4* __restrict__ p, int n4) {
    int i = blockIdx.x * 256 + threadIdx.x;
    if (i < n4) p[i] = make_float4(0.f, 0.f, 0.f, 0.f);
}

// ---------------------------------------------------------------------------
// Edge scatter: agg[dst[e],:] += adj[e] * S[src[e],:].
// 2 edges per 256-thread block; each thread owns 4 columns (float4 gather,
// 4x global_atomic_add_f32).  S (64MB) lives in the 192MB L2, so the 1GB of
// gather traffic stays on-chip; atomics resolve at L2.
// ---------------------------------------------------------------------------
__global__ __launch_bounds__(256) void scatter_kernel(
    const float* __restrict__ S, const int* __restrict__ src,
    const int* __restrict__ dst, const float* __restrict__ w,
    float* __restrict__ agg, int E) {
    int e = blockIdx.x * 2 + (threadIdx.x >> 7);
    if (e >= E) return;
    int c = (threadIdx.x & 127) << 2;
    int s = src[e];
    int d = dst[e];
    float a = w[e];
    float4 v = *(const float4*)(S + (size_t)s * 512 + c);
    float* o = agg + (size_t)d * 512 + c;
    unsafeAtomicAdd(o + 0, a * v.x);
    unsafeAtomicAdd(o + 1, a * v.y);
    unsafeAtomicAdd(o + 2, a * v.z);
    unsafeAtomicAdd(o + 3, a * v.w);
}

// ---------------------------------------------------------------------------
// Per-graph mean over 64 contiguous rows, +b1 (identity act folded), sigmoid.
// One block per graph; thread owns 2 columns.
// ---------------------------------------------------------------------------
__global__ __launch_bounds__(256) void pool_kernel(
    const float* __restrict__ agg, const float* __restrict__ b,
    float* __restrict__ pooled) {
    int g = blockIdx.x;
    int c = threadIdx.x << 1;
    const float* base = agg + (size_t)g * 64 * 512 + c;
    float s0 = 0.f, s1 = 0.f;
    for (int r = 0; r < 64; ++r) {
        s0 += base[(size_t)r * 512 + 0];
        s1 += base[(size_t)r * 512 + 1];
    }
    float m0 = s0 * (1.f / 64.f) + b[c];
    float m1 = s1 * (1.f / 64.f) + b[c + 1];
    pooled[(size_t)g * 512 + c]     = 1.f / (1.f + __expf(-m0));
    pooled[(size_t)g * 512 + c + 1] = 1.f / (1.f + __expf(-m1));
}

// ---------------------------------------------------------------------------
// Repeat-expand [G,512] -> [N,512]; graph_ids = node>>6 (64 nodes per graph).
// ---------------------------------------------------------------------------
__global__ __launch_bounds__(256) void expand_kernel(
    const float* __restrict__ y, float* __restrict__ out) {
    int i   = blockIdx.x * 256 + threadIdx.x;   // over N*128 float4s
    int row = i >> 7;
    int c4  = i & 127;
    ((float4*)out)[i] = ((const float4*)y)[(size_t)((row >> 6) << 7) + c4];
}

// ---------------------------------------------------------------------------
extern "C" void kernel_launch(void* const* d_in, const int* in_sizes, int n_in,
                              void* d_out, int out_size, void* d_ws, size_t ws_size,
                              hipStream_t stream) {
    const float* feat = (const float*)d_in[0];
    const int*   src  = (const int*)d_in[1];
    const int*   dst  = (const int*)d_in[2];
    const float* adj  = (const float*)d_in[3];
    /* d_in[4] graph_ids: arange(N)//64, computed as n>>6 on device */
    const float* W0  = (const float*)d_in[5];
    const float* b0  = (const float*)d_in[6];
    const float* W1  = (const float*)d_in[7];
    const float* b1  = (const float*)d_in[8];
    const float* gW1 = (const float*)d_in[9];
    const float* gb1 = (const float*)d_in[10];
    const float* gW2 = (const float*)d_in[11];
    const float* gb2 = (const float*)d_in[12];
    const float* gW3 = (const float*)d_in[13];
    const float* gb3 = (const float*)d_in[14];
    const float* gWs = (const float*)d_in[15];
    const float* gbs = (const float*)d_in[16];
    float* out = (float*)d_out;

    const int N = 32768, D = 512, E = 524288, G = 512;

    char* ws = (char*)d_ws;
    size_t off = 0;
    auto take = [&](size_t bytes) -> char* {
        char* p = ws + off;
        off += (bytes + 255) & ~(size_t)255;
        return p;
    };
    uint16_t* Apack  = (uint16_t*)take((size_t)N * D * 2);   // 32 MB
    float*    agg    = (float*)take((size_t)N * D * 4);      // 64 MB
    float*    pooled = (float*)take((size_t)G * D * 4);
    uint16_t* Gpack  = (uint16_t*)take((size_t)G * D * 2);
    float*    z      = (float*)take((size_t)G * D * 4);
    float*    z3     = (float*)take((size_t)G * D * 4);
    float*    yv     = (float*)take((size_t)G * D * 4);
    uint16_t* Zpack  = (uint16_t*)take((size_t)G * D * 2);
    uint16_t* Bp[6];
    for (int i = 0; i < 6; ++i) Bp[i] = (uint16_t*)take((size_t)D * D * 2);

    float* S = out;   // support matrix scratch; dead before final expand

    dim3 blk(256);
    const int packA_big_blocks = (N / 16) * 16 * 32 / 256;   // 4096
    const int packA_sml_blocks = (G / 16) * 16 * 32 / 256;   // 64
    const int zero_blocks      = N * D / 4 / 256;            // 16384
    dim3 gemm_big(N / 128, D / 64);
    dim3 gemm_sml(G / 128, D / 64);

    // Weight packing (tiny, reused by every GEMM block)
    pack_b_kernel<<<64, blk, 0, stream>>>(W0,  Bp[0]);
    pack_b_kernel<<<64, blk, 0, stream>>>(W1,  Bp[1]);
    pack_b_kernel<<<64, blk, 0, stream>>>(gW1, Bp[2]);
    pack_b_kernel<<<64, blk, 0, stream>>>(gW2, Bp[3]);
    pack_b_kernel<<<64, blk, 0, stream>>>(gW3, Bp[4]);
    pack_b_kernel<<<64, blk, 0, stream>>>(gWs, Bp[5]);

    // --- GCN layer 1: S = feat @ W0 ; agg = A_hat @ S ; h1 = relu(agg+b0) ---
    pack_a_kernel<<<packA_big_blocks, blk, 0, stream>>>(feat, Apack, nullptr, 0, N / 16);
    gemm_bf16_wmma<<<gemm_big, blk, 0, stream>>>(Apack, Bp[0], S, nullptr, nullptr, 0, D);
    zero_kernel<<<zero_blocks, blk, 0, stream>>>((float4*)agg, N * D / 4);
    scatter_kernel<<<E / 2, blk, 0, stream>>>(S, src, dst, adj, agg, E);
    pack_a_kernel<<<packA_big_blocks, blk, 0, stream>>>(agg, Apack, b0, 1, N / 16);

    // --- GCN layer 2: S = h1 @ W1 ; agg = A_hat @ S ; h2 = agg + b1 ---------
    gemm_bf16_wmma<<<gemm_big, blk, 0, stream>>>(Apack, Bp[1], S, nullptr, nullptr, 0, D);
    zero_kernel<<<zero_blocks, blk, 0, stream>>>((float4*)agg, N * D / 4);
    scatter_kernel<<<E / 2, blk, 0, stream>>>(S, src, dst, adj, agg, E);

    // --- Pool: g = sigmoid(mean_64(h2)) on [G,D] --------------------------
    pool_kernel<<<G, blk, 0, stream>>>(agg, b1, pooled);
    pack_a_kernel<<<packA_sml_blocks, blk, 0, stream>>>(pooled, Gpack, nullptr, 0, G / 16);

    // --- FF on [G,D]: 3x(Linear+ReLU) + linear shortcut -------------------
    gemm_bf16_wmma<<<gemm_sml, blk, 0, stream>>>(Gpack, Bp[2], z, gb1, nullptr, 1, D);
    pack_a_kernel<<<packA_sml_blocks, blk, 0, stream>>>(z, Zpack, nullptr, 0, G / 16);
    gemm_bf16_wmma<<<gemm_sml, blk, 0, stream>>>(Zpack, Bp[3], z, gb2, nullptr, 1, D);
    pack_a_kernel<<<packA_sml_blocks, blk, 0, stream>>>(z, Zpack, nullptr, 0, G / 16);
    gemm_bf16_wmma<<<gemm_sml, blk, 0, stream>>>(Zpack, Bp[4], z3, gb3, nullptr, 1, D);
    gemm_bf16_wmma<<<gemm_sml, blk, 0, stream>>>(Gpack, Bp[5], yv, gbs, z3, 0, D);

    // --- Expand [G,D] -> [N,D] --------------------------------------------
    expand_kernel<<<N * D / 4 / 256, blk, 0, stream>>>(yv, out);
}
